// CayleyLinear_24180665877093
// MI455X (gfx1250) — matrix operand order repaired and checked by reading, not compiled, not run on hardware
//
#include <hip/hip_runtime.h>
#include <hip/hip_bf16.h>

// ---------------------------------------------------------------------------
// CayleyLinear on MI455X (gfx1250, wave32):
//   A = skew(angles); Q = inv(I-A)(I+A) via Newton-Schulz (all GEMMs);
//   out = x @ Q + bias.
// GEMM: bf16x3 split (hi*hi + hi*lo + lo*hi) on v_wmma_f32_16x16x32_bf16,
// double-buffered LDS; B operand pre-split to transposed bf16 planes and
// staged with async global->LDS copies (ASYNCcnt / s_wait_asynccnt).
// ---------------------------------------------------------------------------

typedef __attribute__((ext_vector_type(16))) __bf16 v16bf;
typedef __attribute__((ext_vector_type(8)))  __bf16 v8bf;
typedef __attribute__((ext_vector_type(8)))  float  v8f;
typedef __attribute__((ext_vector_type(4)))  float  v4f;

static constexpr int kDim = 1024;   // Q is kDim x kDim
static constexpr int BM = 128;      // block tile M
static constexpr int BN = 128;      // block tile N
static constexpr int BK = 32;       // K step (one bf16 WMMA K)
static constexpr int LDT = 40;      // padded LDS stride (elements) -> 80B rows, 16B aligned

union Frag { v16bf v; v8bf h[2]; };

// ---------------------------------------------------------------------------
// Build P = I + A and Mm = I - A from packed upper-triangle angles.
// ---------------------------------------------------------------------------
__global__ __launch_bounds__(256) void cayley_build_pm(
    const float* __restrict__ angles, float* __restrict__ P, float* __restrict__ Mm)
{
    int g = blockIdx.x * blockDim.x + threadIdx.x;
    if (g >= kDim * kDim) return;
    int i = g >> 10;
    int j = g & (kDim - 1);
    float a = 0.0f;
    if (j > i) {
        int off = i * (kDim - 1) - (i * (i - 1)) / 2;
        a = angles[off + (j - i - 1)];
    } else if (i > j) {
        int off = j * (kDim - 1) - (j * (j - 1)) / 2;
        a = -angles[off + (i - j - 1)];
    }
    float d = (i == j) ? 1.0f : 0.0f;
    P[g]  = d + a;
    Mm[g] = d - a;
}

// ---------------------------------------------------------------------------
// Split S (Kd x Nd f32, row-major) into TRANSPOSED bf16 hi/lo planes
// (Nd x Kd), tiled through LDS so both read and write are coalesced.
// ---------------------------------------------------------------------------
__global__ __launch_bounds__(256) void split_transpose_bf16(
    const float* __restrict__ S, __bf16* __restrict__ Thi, __bf16* __restrict__ Tlo,
    int Kd, int Nd)
{
    __shared__ float tile[32][33];
    int kb = blockIdx.y * 32, nb = blockIdx.x * 32;
    int t = threadIdx.x;
#pragma unroll
    for (int i = 0; i < 4; ++i) {
        int e = t + 256 * i;
        int k = e >> 5, n = e & 31;
        tile[k][n] = S[(size_t)(kb + k) * Nd + nb + n];
    }
    __syncthreads();
#pragma unroll
    for (int i = 0; i < 4; ++i) {
        int e = t + 256 * i;
        int n = e >> 5, k = e & 31;
        float  x = tile[k][n];
        __bf16 h = (__bf16)x;
        __bf16 l = (__bf16)(x - (float)h);
        Thi[(size_t)(nb + n) * Kd + kb + k] = h;
        Tlo[(size_t)(nb + n) * Kd + kb + k] = l;
    }
}

// ---------------------------------------------------------------------------
// D = alpha * (A @ B) + beta * C + bias, with B given as transposed bf16
// hi/lo planes (Nd x Kd). A is f32 (split to bf16 hi/lo in flight).
// 256 threads = 8 waves; each wave computes a 64x32 sub-tile; double-buffered
// LDS; B tiles staged by async global->LDS copies.
// Requires Md%128==0, Nd%128==0, Kd%32==0 (true for all calls here).
// ---------------------------------------------------------------------------
__global__ __launch_bounds__(256) void gemm_bf16x3(
    const float* __restrict__ A,
    const __bf16* __restrict__ Bthi, const __bf16* __restrict__ Btlo,
    float* __restrict__ D, const float* __restrict__ Cmat,
    const float* __restrict__ bias,
    int Md, int Nd, int Kd, float alpha, float beta)
{
    __shared__ __align__(16) __bf16 As_hi[2][BM * LDT];
    __shared__ __align__(16) __bf16 As_lo[2][BM * LDT];
    __shared__ __align__(16) __bf16 Bs_hi[2][BN * LDT];   // [n][k]
    __shared__ __align__(16) __bf16 Bs_lo[2][BN * LDT];

    const int tid   = threadIdx.x;
    const int lane  = tid & 31;
    const int wave  = tid >> 5;
    const int wr    = wave >> 2;        // 0..1 (64 rows each)
    const int wc    = wave & 3;         // 0..3 (32 cols each)
    const int lhalf = lane >> 4;
    const int l15   = lane & 15;
    const int brow  = blockIdx.y * BM;
    const int bcol  = blockIdx.x * BN;

    v8f acc[4][2];
#pragma unroll
    for (int rs = 0; rs < 4; ++rs)
#pragma unroll
        for (int cs = 0; cs < 2; ++cs)
            acc[rs][cs] = (v8f)0.0f;

    v4f areg[4];   // in-flight A tile (f32)

    auto loadA = [&](int k0) {
#pragma unroll
        for (int i = 0; i < 4; ++i) {
            int f4 = tid + 256 * i;             // 1024 float4s: 128 rows x 8
            int r  = f4 >> 3;
            int c4 = (f4 & 7) * 4;
            areg[i] = *(const v4f*)(A + (size_t)(brow + r) * Kd + k0 + c4);
        }
    };
    auto storeA = [&](int buf) {
#pragma unroll
        for (int i = 0; i < 4; ++i) {
            int f4 = tid + 256 * i;
            int r  = f4 >> 3;
            int c4 = (f4 & 7) * 4;
#pragma unroll
            for (int j = 0; j < 4; ++j) {
                float  x = areg[i][j];
                __bf16 h = (__bf16)x;
                __bf16 l = (__bf16)(x - (float)h);
                As_hi[buf][r * LDT + c4 + j] = h;
                As_lo[buf][r * LDT + c4 + j] = l;
            }
        }
    };
    // Async copy of one B tile (both bf16 planes, 128 rows x 32 k each) into
    // LDS. 16B per lane per instruction; i=0,1 -> hi plane, i=2,3 -> lo plane.
    auto asyncB = [&](int buf, int k0) {
#pragma unroll
        for (int i = 0; i < 4; ++i) {
            const __bf16* srcp = (i < 2) ? Bthi : Btlo;
            __bf16* dstp = (i < 2) ? &Bs_hi[buf][0] : &Bs_lo[buf][0];
            int cc = tid + 256 * (i & 1);       // 0..511 chunk within plane
            int n  = cc >> 2;                   // 0..127
            int q  = cc & 3;                    // 16B chunk within row
            unsigned long long ga =
                (unsigned long long)(uintptr_t)(srcp + (size_t)(bcol + n) * Kd + k0 + q * 8);
            unsigned lofs = (unsigned)(uintptr_t)(dstp + n * LDT + q * 8);
            asm volatile("global_load_async_to_lds_b128 %0, %1, off"
                         :: "v"(lofs), "v"(ga) : "memory");
        }
    };
    auto compute = [&](int buf) {
        Frag ahi[4], alo[4], bhi[2], blo[2];
        // A (16-bit 16x32): lane-half 0 -> K {0..7,16..23}, half 1 -> {8..15,24..31}
        const int ka = lhalf ? 8 : 0;
#pragma unroll
        for (int rs = 0; rs < 4; ++rs) {
            const __bf16* ph = &As_hi[buf][(wr * 64 + rs * 16 + l15) * LDT + ka];
            ahi[rs].h[0] = *(const v8bf*)ph;
            ahi[rs].h[1] = *(const v8bf*)(ph + 16);
            const __bf16* pl = &As_lo[buf][(wr * 64 + rs * 16 + l15) * LDT + ka];
            alo[rs].h[0] = *(const v8bf*)pl;
            alo[rs].h[1] = *(const v8bf*)(pl + 16);
        }
        // B (16-bit 32x16): lane-half 0 -> K {0..7,16..23}, half 1 -> {8..15,24..31}
        const int kb = lhalf * 8;
#pragma unroll
        for (int cs = 0; cs < 2; ++cs) {
            const __bf16* ph = &Bs_hi[buf][(wc * 32 + cs * 16 + l15) * LDT + kb];
            bhi[cs].h[0] = *(const v8bf*)ph;
            bhi[cs].h[1] = *(const v8bf*)(ph + 16);
            const __bf16* pl = &Bs_lo[buf][(wc * 32 + cs * 16 + l15) * LDT + kb];
            blo[cs].h[0] = *(const v8bf*)pl;
            blo[cs].h[1] = *(const v8bf*)(pl + 16);
        }
#pragma unroll
        for (int rs = 0; rs < 4; ++rs)
#pragma unroll
            for (int cs = 0; cs < 2; ++cs) {
                acc[rs][cs] = __builtin_amdgcn_wmma_f32_16x16x32_bf16(
                    false, ahi[rs].v, false, bhi[cs].v, (short)0, acc[rs][cs], false, false);
                acc[rs][cs] = __builtin_amdgcn_wmma_f32_16x16x32_bf16(
                    false, ahi[rs].v, false, blo[cs].v, (short)0, acc[rs][cs], false, false);
                acc[rs][cs] = __builtin_amdgcn_wmma_f32_16x16x32_bf16(
                    false, alo[rs].v, false, bhi[cs].v, (short)0, acc[rs][cs], false, false);
            }
    };

    // ---- prologue: stage tile 0 into buffer 0 ----
    loadA(0);
    asyncB(0, 0);
    storeA(0);
    asm volatile("s_wait_asynccnt 0x0" ::: "memory");
    __syncthreads();

    // ---- software-pipelined main loop ----
    const int nk = Kd / BK;
    for (int kt = 0; kt < nk; ++kt) {
        const int buf = kt & 1;
        const int nxt = buf ^ 1;
        const bool more = (kt + 1 < nk);
        if (more) {
            loadA((kt + 1) * BK);       // global f32 loads overlap compute
            asyncB(nxt, (kt + 1) * BK); // async engine fills next B buffer
        }
        compute(buf);                   // 24 WMMAs on current buffer
        if (more) {
            storeA(nxt);                // convert+store A after compute
            asm volatile("s_wait_asynccnt 0x0" ::: "memory");
        }
        __syncthreads();
    }

    // ---- epilogue: D = alpha*acc + beta*C + bias ----
#pragma unroll
    for (int rs = 0; rs < 4; ++rs)
#pragma unroll
        for (int cs = 0; cs < 2; ++cs) {
            int r0 = brow + wr * 64 + rs * 16 + lhalf * 8;
            int c  = bcol + wc * 32 + cs * 16 + l15;
            float bv = (bias != nullptr) ? bias[c] : 0.0f;
#pragma unroll
            for (int v = 0; v < 8; ++v) {
                float val = alpha * acc[rs][cs][v] + bv;
                if (beta != 0.0f)
                    val += beta * Cmat[(size_t)(r0 + v) * Nd + c];
                D[(size_t)(r0 + v) * Nd + c] = val;
            }
        }
}

// ---------------------------------------------------------------------------
// Host-side orchestration (all on `stream`, graph-capture safe).
// ---------------------------------------------------------------------------
extern "C" void kernel_launch(void* const* d_in, const int* in_sizes, int n_in,
                              void* d_out, int out_size, void* d_ws, size_t ws_size,
                              hipStream_t stream) {
    const float* x      = (const float*)d_in[0];   // [8*4096, 1024]
    const float* angles = (const float*)d_in[1];   // [1024*1023/2]
    const float* bias   = (const float*)d_in[2];   // [1024]
    float* out = (float*)d_out;
    float* ws  = (float*)d_ws;

    const size_t MM = (size_t)kDim * kDim;         // 1M floats = 4 MB
    float* P  = ws + 0 * MM;   // I + A
    float* Mm = ws + 1 * MM;   // I - A
    float* T  = ws + 2 * MM;   // scratch (Mm @ X)
    float* Xa = ws + 3 * MM;   // Newton ping
    float* Xb = ws + 4 * MM;   // Newton pong
    float* Q  = ws + 5 * MM;   // final rotation
    __bf16* Bthi = (__bf16*)(ws + 6 * MM);         // transposed bf16 planes (2MB each)
    __bf16* Btlo = Bthi + MM;

    // 1) Build P = I+A, Mm = I-A.
    cayley_build_pm<<<(kDim * kDim) / 256, 256, 0, stream>>>(angles, P, Mm);

    dim3 sgrid(kDim / 32, kDim / 32);
    auto split = [&](const float* S) {
        split_transpose_bf16<<<sgrid, 256, 0, stream>>>(S, Bthi, Btlo, kDim, kDim);
    };
    auto gemm = [&](const float* Am, const float* Bm, float* Dst,
                    const float* Cm, const float* bv, int Md, float al, float be) {
        split(Bm);                                  // pre-split B operand
        dim3 grid(kDim / BN, Md / BM);
        gemm_bf16x3<<<grid, 256, 0, stream>>>(Am, Bthi, Btlo, Dst, Cm, bv,
                                              Md, kDim, kDim, al, be);
    };

    // 2) Newton-Schulz: X_{k+1} = X_k (2I - Mm X_k), starting X1 = P = I+A.
    //    ||A||2 ~ 0.64 -> residual after 6 iterations ~ 1e-13 (quadratic).
    const float* Xin = P;
    float* Xout = Xa;
    for (int it = 0; it < 6; ++it) {
        gemm(Mm, Xin, T, nullptr, nullptr, kDim, 1.0f, 0.0f);   // T  = Mm @ X
        gemm(Xin, T, Xout, Xin, nullptr, kDim, -1.0f, 2.0f);    // X' = 2X - X @ T
        Xin  = Xout;
        Xout = (Xout == Xa) ? Xb : Xa;
    }

    // 3) Q = X @ P = inv(I-A)(I+A).
    gemm(Xin, P, Q, nullptr, nullptr, kDim, 1.0f, 0.0f);

    // 4) out = x @ Q + bias  (M = 32768; Q's bf16 planes stay L2-resident).
    const int Mbig = in_sizes[0] / kDim;
    split(Q);
    dim3 gridB(kDim / BN, Mbig / BM);
    gemm_bf16x3<<<gridB, 256, 0, stream>>>(x, Bthi, Btlo, out, nullptr, bias,
                                           Mbig, kDim, kDim, 1.0f, 0.0f);
}
